// RelativeMultiHeadAttention_16484084483016
// MI455X (gfx1250) — compile-verified, matrix operand order checked by reading, not run on hardware
//
#include <hip/hip_runtime.h>
#include <hip/hip_bf16.h>

// Problem constants (match reference setup_inputs)
constexpr int Bn = 8;
constexpr int Cn = 512;
constexpr int Ln = 1024;
constexpr int Hn = 8;
constexpr int DhN = 64;
constexpr int WIN = 4;          // window
constexpr int NT = 9;           // 2*WIN+1 relative positions

typedef __bf16 v8bf  __attribute__((ext_vector_type(8)));
typedef __bf16 v16bf __attribute__((ext_vector_type(16)));
typedef float  v8f   __attribute__((ext_vector_type(8)));

__device__ __forceinline__ v8bf ld8(const __bf16* p) { return *(const v8bf*)p; }
__device__ __forceinline__ v16bf cat16(v8bf a, v8bf b) {
  return __builtin_shufflevector(a, b, 0,1,2,3,4,5,6,7,8,9,10,11,12,13,14,15);
}
__device__ __forceinline__ v8f wmma_bf16(v16bf a, v16bf b, v8f c) {
  // D = A(16x32 bf16) * B(32x16 bf16) + C(f32)
  return __builtin_amdgcn_wmma_f32_16x16x32_bf16(false, a, false, b, (short)0, c, false, false);
}

// ---------------------------------------------------------------------------
// Kernel 1: LDS-tiled transpose + downconvert: x (B,C,L) f32 -> xT (B,L,C) bf16
// ---------------------------------------------------------------------------
__global__ void k_transpose_bf16(const float* __restrict__ x, __bf16* __restrict__ xT) {
  __shared__ float tile[32][33];
  int b  = blockIdx.z;
  int c0 = blockIdx.y * 32;
  int l0 = blockIdx.x * 32;
  const float* xp = x + (size_t)b * Cn * Ln;
  #pragma unroll
  for (int i = 0; i < 4; ++i) {
    int c = c0 + threadIdx.y + i * 8;
    int l = l0 + threadIdx.x;
    tile[threadIdx.y + i * 8][threadIdx.x] = xp[(size_t)c * Ln + l];
  }
  __syncthreads();
  __bf16* xtp = xT + (size_t)b * Ln * Cn;
  #pragma unroll
  for (int i = 0; i < 4; ++i) {
    int l = l0 + threadIdx.y + i * 8;
    int c = c0 + threadIdx.x;
    xtp[(size_t)l * Cn + c] = (__bf16)tile[threadIdx.x][threadIdx.y + i * 8];
  }
}

// ---------------------------------------------------------------------------
// Kernel 2: f32 -> bf16 weight convert
// ---------------------------------------------------------------------------
__global__ void k_f32_to_bf16(const float* __restrict__ in, __bf16* __restrict__ out, int n) {
  int i = blockIdx.x * blockDim.x + threadIdx.x;
  if (i < n) out[i] = (__bf16)in[i];
}

// ---------------------------------------------------------------------------
// Kernel 3: QKV projections. One wave = 16(l) x 64(c) output (4 accumulators),
//           reusing the xT-row operand across the 4 c-subtiles. K-loop over C.
//   p=0 (q), p=1 (k): YT[l,c] = sum_k xT[l,k]*W[c,k]  -> stored (B,L,C), q scaled
//   p=2 (v):          Y[c,l]  = sum_k W[c,k]*xT[l,k]  -> stored (B,C,L)
// ---------------------------------------------------------------------------
__global__ void k_qkv_proj(const __bf16* __restrict__ xT, const __bf16* __restrict__ Wbf,
                           const float* __restrict__ bq, const float* __restrict__ bk,
                           const float* __restrict__ bv,
                           __bf16* __restrict__ qT, __bf16* __restrict__ kT,
                           __bf16* __restrict__ vS) {
  int b = blockIdx.z / 3, p = blockIdx.z % 3;
  int l0 = blockIdx.x * 16, c0 = blockIdx.y * 64;
  int lane = threadIdx.x;
  int lo = lane & 15, hi = lane >> 4, koff = hi * 8;

  const __bf16* W    = Wbf + (size_t)p * Cn * Cn;
  const __bf16* xrow = xT + (size_t)b * Ln * Cn + (size_t)(l0 + lo) * Cn;
  const __bf16* wrow[4];
  #pragma unroll
  for (int t = 0; t < 4; ++t) wrow[t] = W + (size_t)(c0 + t * 16 + lo) * Cn;

  v8f acc[4] = {{}, {}, {}, {}};
  if (p < 2) {
    for (int k0 = 0; k0 < Cn; k0 += 32) {
      v16bf a = cat16(ld8(xrow + k0 + koff), ld8(xrow + k0 + 16 + koff));
      #pragma unroll
      for (int t = 0; t < 4; ++t) {
        v16bf bm = cat16(ld8(wrow[t] + k0 + koff), ld8(wrow[t] + k0 + 16 + koff));
        acc[t] = wmma_bf16(a, bm, acc[t]);
      }
    }
    const float* bias = (p == 0) ? bq : bk;
    float scale = (p == 0) ? 0.125f : 1.0f;   // Dh^-0.5 = 1/8
    __bf16* out = ((p == 0) ? qT : kT) + (size_t)b * Ln * Cn;
    #pragma unroll
    for (int t = 0; t < 4; ++t) {
      float bl = bias[c0 + t * 16 + lo];
      #pragma unroll
      for (int r = 0; r < 8; ++r) {
        float v = (acc[t][r] + bl) * scale;
        out[(size_t)(l0 + r + 8 * hi) * Cn + c0 + t * 16 + lo] = (__bf16)v;
      }
    }
  } else {
    for (int k0 = 0; k0 < Cn; k0 += 32) {
      v16bf bm = cat16(ld8(xrow + k0 + koff), ld8(xrow + k0 + 16 + koff));
      #pragma unroll
      for (int t = 0; t < 4; ++t) {
        v16bf a = cat16(ld8(wrow[t] + k0 + koff), ld8(wrow[t] + k0 + 16 + koff));
        acc[t] = wmma_bf16(a, bm, acc[t]);
      }
    }
    __bf16* out = vS + (size_t)b * Cn * Ln;
    #pragma unroll
    for (int t = 0; t < 4; ++t) {
      #pragma unroll
      for (int r = 0; r < 8; ++r) {
        int c = c0 + t * 16 + r + 8 * hi;
        out[(size_t)c * Ln + l0 + lo] = (__bf16)(acc[t][r] + bv[c]);
      }
    }
  }
}

// ---------------------------------------------------------------------------
// Kernel 4: rel_q[b,h,l,t] = dot(q_scaled[b,h,l,:], emb_rel_k[t,:])  (stride-16 pad)
// ---------------------------------------------------------------------------
__global__ void k_relq(const __bf16* __restrict__ qT, const float* __restrict__ emb_rel_k,
                       float* __restrict__ relq) {
  __shared__ float relk[NT * DhN];
  for (int i = threadIdx.x; i < NT * DhN; i += 256) relk[i] = emb_rel_k[i];
  __syncthreads();
  int idx = blockIdx.x * 256 + threadIdx.x;   // (b*H + h)*L + l
  int l  = idx % Ln;
  int bh = idx / Ln;
  int h  = bh % Hn, b = bh / Hn;
  const __bf16* q = qT + ((size_t)b * Ln + l) * Cn + h * DhN;
  float qf[DhN];
  #pragma unroll
  for (int d = 0; d < DhN; ++d) qf[d] = (float)q[d];
  #pragma unroll
  for (int t = 0; t < NT; ++t) {
    float a = 0.f;
    #pragma unroll
    for (int d = 0; d < DhN; ++d) a += qf[d] * relk[t * DhN + d];
    relq[(size_t)idx * 16 + t] = a;
  }
}

// ---------------------------------------------------------------------------
// Kernel 5: flash attention per (b,h,16-row tile); one wave; online softmax;
//           banded relative-K added into S, banded relative-V applied at finalize.
// ---------------------------------------------------------------------------
__global__ void k_attn(const __bf16* __restrict__ qT, const __bf16* __restrict__ kT,
                       const __bf16* __restrict__ vS, const float* __restrict__ relq,
                       const float* __restrict__ emb_rel_v, __bf16* __restrict__ attnT) {
  __shared__ float  sband[16][12];
  __shared__ float  relv[NT * DhN];
  __shared__ __bf16 pbuf[16][32];

  int lane = threadIdx.x;
  int l0 = blockIdx.x * 16;
  int h = blockIdx.y, b = blockIdx.z;

  for (int i = lane; i < 16 * 12; i += 32) ((float*)sband)[i] = -__builtin_inff();
  for (int i = lane; i < NT * DhN; i += 32) relv[i] = emb_rel_v[i];
  __syncthreads();

  int lo = lane & 15, hi = lane >> 4, koff = hi * 8;

  const __bf16* qrow = qT + ((size_t)b * Ln + l0 + lo) * Cn + h * DhN;
  v16bf qa0 = cat16(ld8(qrow + koff),      ld8(qrow + 16 + koff));
  v16bf qa1 = cat16(ld8(qrow + 32 + koff), ld8(qrow + 48 + koff));

  const __bf16* kbase = kT + (size_t)b * Ln * Cn + h * DhN;
  const __bf16* vbase = vS + (size_t)b * Cn * Ln + (size_t)(h * DhN) * Ln;
  const float*  rq    = relq + (size_t)(b * Hn + h) * Ln * 16;

  v8f o[4] = {{}, {}, {}, {}};
  float mrow[8], zrow[8];
  #pragma unroll
  for (int r = 0; r < 8; ++r) { mrow[r] = -__builtin_inff(); zrow[r] = 0.f; }

  for (int j0 = 0; j0 < Ln; j0 += 32) {
    // ---- S = q * k^T for 16x32 chunk (4 WMMAs) ----
    v8f s[2];
    #pragma unroll
    for (int t = 0; t < 2; ++t) {
      const __bf16* krow = kbase + (size_t)(j0 + t * 16 + lo) * Cn;
      v16bf b0 = cat16(ld8(krow + koff),      ld8(krow + 16 + koff));
      v16bf b1 = cat16(ld8(krow + 32 + koff), ld8(krow + 48 + koff));
      v8f acc = {};
      acc = wmma_bf16(qa0, b0, acc);
      acc = wmma_bf16(qa1, b1, acc);
      s[t] = acc;
    }
    // ---- banded relative-K add + band capture + chunk row-max ----
    float cmax[8];
    #pragma unroll
    for (int r = 0; r < 8; ++r) {
      int l = l0 + r + 8 * hi;
      float v0 = s[0][r], v1 = s[1][r];
      int d0 = (j0 + lo) - l, d1 = (j0 + 16 + lo) - l;
      if (d0 >= -WIN && d0 <= WIN) { v0 += rq[(size_t)l * 16 + d0 + WIN]; sband[r + 8 * hi][d0 + WIN] = v0; }
      if (d1 >= -WIN && d1 <= WIN) { v1 += rq[(size_t)l * 16 + d1 + WIN]; sband[r + 8 * hi][d1 + WIN] = v1; }
      s[0][r] = v0; s[1][r] = v1;
      cmax[r] = fmaxf(v0, v1);
    }
    #pragma unroll
    for (int m = 1; m < 16; m <<= 1)
      #pragma unroll
      for (int r = 0; r < 8; ++r)
        cmax[r] = fmaxf(cmax[r], __shfl_xor(cmax[r], m, 32));
    // ---- online softmax update ----
    float corr[8], mnew[8], psum[8];
    #pragma unroll
    for (int r = 0; r < 8; ++r) {
      mnew[r] = fmaxf(mrow[r], cmax[r]);
      corr[r] = __expf(mrow[r] - mnew[r]);
      mrow[r] = mnew[r];
      float p0 = __expf(s[0][r] - mnew[r]);
      float p1 = __expf(s[1][r] - mnew[r]);
      psum[r] = p0 + p1;
      pbuf[r + 8 * hi][lo]      = (__bf16)p0;
      pbuf[r + 8 * hi][16 + lo] = (__bf16)p1;
    }
    #pragma unroll
    for (int m = 1; m < 16; m <<= 1)
      #pragma unroll
      for (int r = 0; r < 8; ++r)
        psum[r] += __shfl_xor(psum[r], m, 32);
    #pragma unroll
    for (int r = 0; r < 8; ++r) zrow[r] = zrow[r] * corr[r] + psum[r];
    #pragma unroll
    for (int dt = 0; dt < 4; ++dt)
      #pragma unroll
      for (int r = 0; r < 8; ++r)
        o[dt][r] *= corr[r];
    __syncthreads();
    // ---- O += P(16x32) * V(32x64): re-read P in A-operand layout from LDS ----
    v16bf pa = cat16(*(const v8bf*)&pbuf[lo][koff], *(const v8bf*)&pbuf[lo][16 + koff]);
    #pragma unroll
    for (int dt = 0; dt < 4; ++dt) {
      const __bf16* vrow = vbase + (size_t)(dt * 16 + lo) * Ln;
      v16bf vb = cat16(ld8(vrow + j0 + koff), ld8(vrow + j0 + 16 + koff));
      o[dt] = wmma_bf16(pa, vb, o[dt]);
    }
    __syncthreads();
  }

  // ---- finalize: 1/Z scaling + banded relative-V contribution ----
  #pragma unroll
  for (int r = 0; r < 8; ++r) {
    int m = r + 8 * hi;
    int l = l0 + m;
    float inv = 1.0f / zrow[r];
    float aw[NT];
    #pragma unroll
    for (int t = 0; t < NT; ++t) aw[t] = __expf(sband[m][t] - mrow[r]) * inv;
    #pragma unroll
    for (int dt = 0; dt < 4; ++dt) {
      int d = dt * 16 + lo;
      float relsum = 0.f;
      #pragma unroll
      for (int t = 0; t < NT; ++t) relsum += aw[t] * relv[t * DhN + d];
      float val = o[dt][r] * inv + relsum;
      attnT[((size_t)b * Ln + l) * Cn + h * DhN + d] = (__bf16)val;
    }
  }
}

// ---------------------------------------------------------------------------
// Kernel 6: output projection  out[b,o,l] = sum_c Wo[o,c]*attnT[b,l,c] + bo[o]
//           One wave = 64(o) x 16(l), reusing attnT-row operand across 4 o-tiles.
// ---------------------------------------------------------------------------
__global__ void k_outproj(const __bf16* __restrict__ Wo_bf, const __bf16* __restrict__ attnT,
                          const float* __restrict__ bo, float* __restrict__ out) {
  int b = blockIdx.z;
  int l0 = blockIdx.x * 16, o0 = blockIdx.y * 64;
  int lane = threadIdx.x;
  int lo = lane & 15, hi = lane >> 4, koff = hi * 8;
  const __bf16* arow[4];
  #pragma unroll
  for (int t = 0; t < 4; ++t) arow[t] = Wo_bf + (size_t)(o0 + t * 16 + lo) * Cn;
  const __bf16* brow = attnT + ((size_t)b * Ln + l0 + lo) * Cn;
  v8f acc[4] = {{}, {}, {}, {}};
  for (int k0 = 0; k0 < Cn; k0 += 32) {
    v16bf bm = cat16(ld8(brow + k0 + koff), ld8(brow + k0 + 16 + koff));
    #pragma unroll
    for (int t = 0; t < 4; ++t) {
      v16bf a = cat16(ld8(arow[t] + k0 + koff), ld8(arow[t] + k0 + 16 + koff));
      acc[t] = wmma_bf16(a, bm, acc[t]);
    }
  }
  #pragma unroll
  for (int t = 0; t < 4; ++t) {
    #pragma unroll
    for (int r = 0; r < 8; ++r) {
      int oc = o0 + t * 16 + r + 8 * hi;
      out[((size_t)b * Cn + oc) * Ln + l0 + lo] = acc[t][r] + bo[oc];
    }
  }
}

// ---------------------------------------------------------------------------
extern "C" void kernel_launch(void* const* d_in, const int* in_sizes, int n_in,
                              void* d_out, int out_size, void* d_ws, size_t ws_size,
                              hipStream_t stream) {
  const float* x   = (const float*)d_in[0];
  const float* Wq  = (const float*)d_in[1];
  const float* bq  = (const float*)d_in[2];
  const float* Wk  = (const float*)d_in[3];
  const float* bk  = (const float*)d_in[4];
  const float* Wv  = (const float*)d_in[5];
  const float* bv  = (const float*)d_in[6];
  const float* Wo  = (const float*)d_in[7];
  const float* bo  = (const float*)d_in[8];
  const float* erk = (const float*)d_in[9];
  const float* erv = (const float*)d_in[10];

  char* w = (char*)d_ws;
  size_t off = 0;
  auto carve = [&](size_t bytes) { char* p = w + off; off += (bytes + 255) & ~(size_t)255; return p; };

  __bf16* xT    = (__bf16*)carve((size_t)Bn * Ln * Cn * 2);
  __bf16* Wbf   = (__bf16*)carve((size_t)4 * Cn * Cn * 2);   // q,k,v,o
  __bf16* qT    = (__bf16*)carve((size_t)Bn * Ln * Cn * 2);
  __bf16* kTb   = (__bf16*)carve((size_t)Bn * Ln * Cn * 2);
  __bf16* vS    = (__bf16*)carve((size_t)Bn * Cn * Ln * 2);
  __bf16* attnT = (__bf16*)carve((size_t)Bn * Ln * Cn * 2);
  float*  relq  = (float*)carve((size_t)Bn * Hn * Ln * 16 * 4);
  (void)ws_size; (void)in_sizes; (void)n_in; (void)out_size;

  // 1) x -> xT bf16
  k_transpose_bf16<<<dim3(Ln / 32, Cn / 32, Bn), dim3(32, 8), 0, stream>>>(x, xT);
  // 2) weights -> bf16
  int nw = Cn * Cn;
  k_f32_to_bf16<<<(nw + 255) / 256, 256, 0, stream>>>(Wq, Wbf + 0 * (size_t)nw, nw);
  k_f32_to_bf16<<<(nw + 255) / 256, 256, 0, stream>>>(Wk, Wbf + 1 * (size_t)nw, nw);
  k_f32_to_bf16<<<(nw + 255) / 256, 256, 0, stream>>>(Wv, Wbf + 2 * (size_t)nw, nw);
  k_f32_to_bf16<<<(nw + 255) / 256, 256, 0, stream>>>(Wo, Wbf + 3 * (size_t)nw, nw);
  // 3) q/k/v projections (WMMA, 16x64 per wave)
  k_qkv_proj<<<dim3(Ln / 16, Cn / 64, Bn * 3), 32, 0, stream>>>(xT, Wbf, bq, bk, bv, qT, kTb, vS);
  // 4) relative-K dot products
  k_relq<<<(Bn * Hn * Ln) / 256, 256, 0, stream>>>(qT, erk, relq);
  // 5) flash attention with relative bias (WMMA)
  k_attn<<<dim3(Ln / 16, Hn, Bn), 32, 0, stream>>>(qT, kTb, vS, relq, erv, attnT);
  // 6) output projection (WMMA, 64x16 per wave)
  k_outproj<<<dim3(Ln / 16, Cn / 64, Bn), 32, 0, stream>>>(Wbf + 3 * (size_t)nw, attnT, bo, (float*)d_out);
}
